// MixtureOfExperts_91001767067920
// MI455X (gfx1250) — matrix operand order repaired and checked by reading, not compiled, not run on hardware
//
#include <hip/hip_runtime.h>
#include <hip/hip_bf16.h>
#include <math.h>

#define BATCHN   2048
#define N_VARS   7
#define NROWS    (BATCHN * N_VARS)   // 14336
#define SEQ      512
#define DM       512
#define NE       8
#define HID      256
#define MAVG     25
#define PADW     12                  // (MAVG-1)/2

typedef _Float16 h16;
typedef __attribute__((ext_vector_type(8)))  _Float16 v8h;
typedef __attribute__((ext_vector_type(16))) _Float16 v16h;
typedef __attribute__((ext_vector_type(8)))  float    v8f;

__device__ __forceinline__ v16h cat16(v8h a, v8h b) {
  return __builtin_shufflevector(a, b, 0,1,2,3,4,5,6,7,8,9,10,11,12,13,14,15);
}

// ---------------------------------------------------------------------------
// Zero output + expert counters
// ---------------------------------------------------------------------------
__global__ void k_zero(float* __restrict__ y, int ntot, int* __restrict__ cnt) {
  int i = blockIdx.x * blockDim.x + threadIdx.x;
  if (i < ntot) y[i] = 0.0f;
  if (i < NE)   cnt[i] = 0;
}

// ---------------------------------------------------------------------------
// Per-row: RevIN normalize + moving-average decomposition; emit f16 streams.
// One block (128 threads) per row.
// ---------------------------------------------------------------------------
__global__ void k_prep(const float* __restrict__ x,
                       const float* __restrict__ rw,
                       const float* __restrict__ rb,
                       h16* __restrict__ Xh,
                       h16* __restrict__ Sh,
                       h16* __restrict__ Th) {
  const int n   = blockIdx.x;
  const int var = n % N_VARS;
  const int tid = threadIdx.x;

  __shared__ float xr[SEQ];
  __shared__ float xn[SEQ];
  __shared__ float red[128];

  float s = 0.0f;
  for (int i = tid; i < SEQ; i += 128) {
    float v = x[(size_t)n * SEQ + i];
    xr[i] = v;
    s += v;
  }
  red[tid] = s;
  __syncthreads();
  for (int st = 64; st > 0; st >>= 1) {
    if (tid < st) red[tid] += red[tid + st];
    __syncthreads();
  }
  const float mean = red[0] * (1.0f / SEQ);
  __syncthreads();

  float ss = 0.0f;
  for (int i = tid; i < SEQ; i += 128) {
    float d = xr[i] - mean;
    ss += d * d;
  }
  red[tid] = ss;
  __syncthreads();
  for (int st = 64; st > 0; st >>= 1) {
    if (tid < st) red[tid] += red[tid + st];
    __syncthreads();
  }
  const float sd  = sqrtf(red[0] * (1.0f / SEQ) + 1e-5f);
  const float w   = rw[var];
  const float b   = rb[var];
  const float inv = 1.0f / sd;

  for (int i = tid; i < SEQ; i += 128) xn[i] = (xr[i] - mean) * inv * w + b;
  __syncthreads();

  for (int i = tid; i < SEQ; i += 128) {
    float acc = 0.0f;
#pragma unroll
    for (int j = -PADW; j <= PADW; ++j) {
      int idx = i + j;
      idx = idx < 0 ? 0 : (idx > SEQ - 1 ? SEQ - 1 : idx);
      acc += xn[idx];
    }
    const float tr = acc * (1.0f / MAVG);
    Xh[(size_t)n * SEQ + i] = (h16)xr[i];
    Th[(size_t)n * SEQ + i] = (h16)tr;
    Sh[(size_t)n * SEQ + i] = (h16)(xn[i] - tr);
  }
}

// ---------------------------------------------------------------------------
// Convert an f32 weight matrix src (P rows x K cols, row-major; logical
// B[k][p] = src[p*K+k]) into WMMA B-fragment-major f16: fragment (kt,pt),
// lane L holds halfs j=0..15 = B[kt*32 + (L>>4)*16 + j][pt*16 + (L&15)],
// stored contiguously so a fragment load is two b128 loads per lane.
// grid.y = expert (src/dst stride P*K).
// ---------------------------------------------------------------------------
__global__ void k_bfrag(const float* __restrict__ src, h16* __restrict__ dst,
                        int P, int K) {
  const int e = blockIdx.y;
  src += (size_t)e * P * K;
  dst += (size_t)e * P * K;

  const int numPt = P >> 4;
  const int total = (K >> 5) * numPt * 32;
  int t = blockIdx.x * blockDim.x + threadIdx.x;
  if (t >= total) return;

  const int L  = t & 31;
  const int f  = t >> 5;
  const int pt = f % numPt;
  const int kt = f / numPt;
  const int prow  = pt * 16 + (L & 15);
  const int kbase = kt * 32 + ((L >> 4) * 16);

  const float* sp = src + (size_t)prow * K + kbase;
  v8h lo, hi;
#pragma unroll
  for (int j = 0; j < 8; ++j) { lo[j] = (h16)sp[j]; hi[j] = (h16)sp[8 + j]; }
  v8h* dp = (v8h*)(dst + (size_t)t * 16);
  dp[0] = lo;
  dp[1] = hi;
}

// ---------------------------------------------------------------------------
// Gating GEMM1: H = relu(Xh @ gw1^T) via WMMA f16 -> f32 -> f16
// block = 256 threads = 8 waves; block tile 16 rows x 128 hidden.
// grid = (HID/128, NROWS/16)
// ---------------------------------------------------------------------------
__global__ void k_gemm1(const h16* __restrict__ Xh,
                        const h16* __restrict__ Bf,
                        h16* __restrict__ H) {
  const int L    = threadIdx.x & 31;
  const int wave = threadIdx.x >> 5;
  const int rowBase = blockIdx.y << 4;
  const int pt = (blockIdx.x << 3) + wave;          // hidden tile 0..15

  const int arow = rowBase + (L & 15);
  const h16* ap  = Xh + (size_t)arow * SEQ + ((L >> 4) * 16);
  const h16* bp  = Bf + ((size_t)pt * 32 + L) * 16; // kt=0 fragment

  v8f c = {};
#pragma unroll
  for (int kt = 0; kt < 16; ++kt) {
    v8h alo = *(const v8h*)(ap + kt * 32);
    v8h ahi = *(const v8h*)(ap + kt * 32 + 8);
    v8h blo = *(const v8h*)(bp);
    v8h bhi = *(const v8h*)(bp + 8);
    bp += 16 * 512;                                  // numPt(16) frags * 512 halfs
    c = __builtin_amdgcn_wmma_f32_16x16x32_f16(false, cat16(alo, ahi),
                                               false, cat16(blo, bhi),
                                               (short)0, c, false, false);
  }

  const int p0  = pt << 4;
  const int rl0 = (L >> 4) * 8;
#pragma unroll
  for (int v = 0; v < 8; ++v) {
    float r = c[v];
    r = r > 0.0f ? r : 0.0f;                         // relu
    H[(size_t)(rowBase + rl0 + v) * HID + p0 + (L & 15)] = (h16)r;
  }
}

// ---------------------------------------------------------------------------
// Gate: logits = H @ gw2^T (8 experts), softmax, top-2 (jax tie order),
// append (row, gate) to per-expert lists. One thread per row.
// ---------------------------------------------------------------------------
__global__ void k_gate(const h16* __restrict__ H,
                       const float* __restrict__ gw2,
                       int* __restrict__ cnt,
                       int* __restrict__ prow,
                       float* __restrict__ pgat,
                       int* __restrict__ reidx,
                       float* __restrict__ rgat) {
  const int n = blockIdx.x * blockDim.x + threadIdx.x;
  if (n >= NROWS) return;

  float acc[NE] = {0.f, 0.f, 0.f, 0.f, 0.f, 0.f, 0.f, 0.f};
  const h16* hp = H + (size_t)n * HID;
  for (int c = 0; c < HID; c += 8) {
    v8h hv = *(const v8h*)(hp + c);
#pragma unroll
    for (int j = 0; j < 8; ++j) {
      const float hj = (float)hv[j];
#pragma unroll
      for (int e = 0; e < NE; ++e) acc[e] += hj * gw2[e * HID + c + j];
    }
  }

  float mx = acc[0];
#pragma unroll
  for (int e = 1; e < NE; ++e) mx = fmaxf(mx, acc[e]);
  float p[NE], sum = 0.0f;
#pragma unroll
  for (int e = 0; e < NE; ++e) { p[e] = __expf(acc[e] - mx); sum += p[e]; }
  const float isum = 1.0f / sum;
#pragma unroll
  for (int e = 0; e < NE; ++e) p[e] *= isum;

  int i0 = 0;
#pragma unroll
  for (int e = 1; e < NE; ++e) if (p[e] > p[i0]) i0 = e;
  int i1 = -1;
#pragma unroll
  for (int e = 0; e < NE; ++e) {
    if (e == i0) continue;
    if (i1 < 0 || p[e] > p[i1]) i1 = e;
  }

  const float v0 = p[i0], v1 = p[i1];
  const float gn = 1.0f / (v0 + v1 + 1e-6f);
  const float g0 = v0 * gn, g1 = v1 * gn;

  int s0 = atomicAdd(&cnt[i0], 1);
  prow[i0 * NROWS + s0] = n;
  pgat[i0 * NROWS + s0] = g0;
  int s1 = atomicAdd(&cnt[i1], 1);
  prow[i1 * NROWS + s1] = n;
  pgat[i1 * NROWS + s1] = g1;

  reidx[2 * n]     = i0;
  reidx[2 * n + 1] = i1;
  rgat[2 * n]      = g0;
  rgat[2 * n + 1]  = g1;
}

// ---------------------------------------------------------------------------
// Expert GEMM: for expert e's gathered rows, c += seasonal.Ws^T + trend.Wt^T
// (two WMMAs per K-step into one accumulator), then y += gate * c (2 FP adds
// per element from zero -> order-independent / deterministic).
// grid = (DM/128, NROWS/16 worst case, NE); block = 256 = 8 waves.
// ---------------------------------------------------------------------------
__global__ void k_moe(const h16* __restrict__ S,
                      const h16* __restrict__ T,
                      const h16* __restrict__ BS,
                      const h16* __restrict__ BT,
                      const int* __restrict__ cnt,
                      const int* __restrict__ prow,
                      const float* __restrict__ pgat,
                      float* __restrict__ y) {
  const int e  = blockIdx.z;
  const int ce = cnt[e];
  const int tileRow = blockIdx.y << 4;
  if (tileRow >= ce) return;

  __shared__ int   rows[16];
  __shared__ float gs[16];
  const int tid = threadIdx.x;
  if (tid < 16) {
    const int s = tileRow + tid;
    if (s < ce) { rows[tid] = prow[e * NROWS + s]; gs[tid] = pgat[e * NROWS + s]; }
    else        { rows[tid] = 0;                   gs[tid] = 0.0f; }
  }
  __syncthreads();

  const int L    = tid & 31;
  const int wave = tid >> 5;
  const int pt   = (blockIdx.x << 3) + wave;       // p tile 0..31

  const h16* bs = BS + (size_t)e * DM * SEQ + ((size_t)pt * 32 + L) * 16;
  const h16* bt = BT + (size_t)e * DM * SEQ + ((size_t)pt * 32 + L) * 16;
  const int arow = rows[L & 15];
  const h16* sp = S + (size_t)arow * SEQ + ((L >> 4) * 16);
  const h16* tp = T + (size_t)arow * SEQ + ((L >> 4) * 16);

  v8f c = {};
  const int kstep = 32 * 512;                       // numPt(32) frags * 512 halfs
#pragma unroll
  for (int kt = 0; kt < 16; ++kt) {
    v8h aslo = *(const v8h*)(sp + kt * 32);
    v8h ashi = *(const v8h*)(sp + kt * 32 + 8);
    v8h atlo = *(const v8h*)(tp + kt * 32);
    v8h athi = *(const v8h*)(tp + kt * 32 + 8);
    v8h bslo = *(const v8h*)(bs);
    v8h bshi = *(const v8h*)(bs + 8);
    v8h btlo = *(const v8h*)(bt);
    v8h bthi = *(const v8h*)(bt + 8);
    bs += kstep;
    bt += kstep;
    c = __builtin_amdgcn_wmma_f32_16x16x32_f16(false, cat16(aslo, ashi),
                                               false, cat16(bslo, bshi),
                                               (short)0, c, false, false);
    c = __builtin_amdgcn_wmma_f32_16x16x32_f16(false, cat16(atlo, athi),
                                               false, cat16(btlo, bthi),
                                               (short)0, c, false, false);
  }

  const int p0  = pt << 4;
  const int rl0 = (L >> 4) * 8;
#pragma unroll
  for (int v = 0; v < 8; ++v) {
    const int rl = rl0 + v;
    if (tileRow + rl < ce) {
      float* dst = &y[(size_t)rows[rl] * DM + p0 + (L & 15)];
      __hip_atomic_fetch_add(dst, gs[rl] * c[v],
                             __ATOMIC_RELAXED, __HIP_MEMORY_SCOPE_AGENT);
    }
  }
}

// ---------------------------------------------------------------------------
// Loss: deterministic importance/load reduction + cv^2. Single block.
// ---------------------------------------------------------------------------
__global__ void k_loss(const int* __restrict__ reidx,
                       const float* __restrict__ rgat,
                       const int* __restrict__ coef,
                       float* __restrict__ out) {
  __shared__ float simp[NE][256];
  __shared__ float sld[NE][256];
  const int tid = threadIdx.x;

  float imp[NE] = {0.f, 0.f, 0.f, 0.f, 0.f, 0.f, 0.f, 0.f};
  float ld[NE]  = {0.f, 0.f, 0.f, 0.f, 0.f, 0.f, 0.f, 0.f};
  for (int n = tid; n < NROWS; n += 256) {
    const int   e0 = reidx[2 * n], e1 = reidx[2 * n + 1];
    const float g0 = rgat[2 * n],  g1 = rgat[2 * n + 1];
#pragma unroll
    for (int e = 0; e < NE; ++e) {
      imp[e] += (e0 == e) ? g0 : 0.0f;
      imp[e] += (e1 == e) ? g1 : 0.0f;
      ld[e]  += (e0 == e && g0 > 0.0f) ? 1.0f : 0.0f;
      ld[e]  += (e1 == e && g1 > 0.0f) ? 1.0f : 0.0f;
    }
  }
#pragma unroll
  for (int e = 0; e < NE; ++e) { simp[e][tid] = imp[e]; sld[e][tid] = ld[e]; }
  __syncthreads();
  for (int s = 128; s > 0; s >>= 1) {
    if (tid < s) {
#pragma unroll
      for (int e = 0; e < NE; ++e) {
        simp[e][tid] += simp[e][tid + s];
        sld[e][tid]  += sld[e][tid + s];
      }
    }
    __syncthreads();
  }
  if (tid == 0) {
    float mi = 0.f, ml = 0.f;
#pragma unroll
    for (int e = 0; e < NE; ++e) { mi += simp[e][0]; ml += sld[e][0]; }
    mi *= (1.0f / NE);
    ml *= (1.0f / NE);
    float vi = 0.f, vl = 0.f;
#pragma unroll
    for (int e = 0; e < NE; ++e) {
      float di = simp[e][0] - mi; vi += di * di;
      float dl = sld[e][0]  - ml; vl += dl * dl;
    }
    vi *= (1.0f / (NE - 1));
    vl *= (1.0f / (NE - 1));
    const float cvi = vi / (mi * mi + 1e-10f);
    const float cvl = vl / (ml * ml + 1e-10f);
    out[(size_t)NROWS * DM] = (cvi + cvl) * (float)coef[0];
  }
}

// ---------------------------------------------------------------------------
// Host launcher
// ---------------------------------------------------------------------------
extern "C" void kernel_launch(void* const* d_in, const int* in_sizes, int n_in,
                              void* d_out, int out_size, void* d_ws, size_t ws_size,
                              hipStream_t stream) {
  (void)in_sizes; (void)n_in; (void)out_size; (void)ws_size;

  const float* x        = (const float*)d_in[0];
  const float* gw1      = (const float*)d_in[1];
  const float* gw2      = (const float*)d_in[2];
  const float* Ws       = (const float*)d_in[3];
  const float* Wt       = (const float*)d_in[4];
  const float* revin_w  = (const float*)d_in[5];
  const float* revin_b  = (const float*)d_in[6];
  const int*   losscoef = (const int*)d_in[7];
  float*       y        = (float*)d_out;

  // workspace carve-up (256-byte aligned regions)
  char* ws = (char*)d_ws;
  size_t off = 0;
  auto carve = [&](size_t bytes) -> char* {
    char* p = ws + off;
    off = (off + bytes + 255) & ~(size_t)255;
    return p;
  };
  h16*   xs_h   = (h16*)  carve((size_t)NROWS * SEQ * 2);
  h16*   seas_h = (h16*)  carve((size_t)NROWS * SEQ * 2);
  h16*   tr_h   = (h16*)  carve((size_t)NROWS * SEQ * 2);
  h16*   Hbuf   = (h16*)  carve((size_t)NROWS * HID * 2);
  h16*   gw1f   = (h16*)  carve((size_t)HID * SEQ * 2);
  h16*   wsf    = (h16*)  carve((size_t)NE * DM * SEQ * 2);
  h16*   wtf    = (h16*)  carve((size_t)NE * DM * SEQ * 2);
  int*   prow   = (int*)  carve((size_t)NE * NROWS * 4);
  float* pgat   = (float*)carve((size_t)NE * NROWS * 4);
  int*   reidx  = (int*)  carve((size_t)NROWS * 2 * 4);
  float* rgat   = (float*)carve((size_t)NROWS * 2 * 4);
  int*   cnt    = (int*)  carve(64);

  const int ytot = NROWS * DM + 1;
  k_zero<<<(ytot + 255) / 256, 256, 0, stream>>>(y, ytot, cnt);

  k_prep<<<NROWS, 128, 0, stream>>>(x, revin_w, revin_b, xs_h, seas_h, tr_h);

  {
    const int thr1 = (SEQ / 32) * (HID / 16) * 32;   // 8192
    k_bfrag<<<dim3((thr1 + 255) / 256, 1), 256, 0, stream>>>(gw1, gw1f, HID, SEQ);
    const int thr2 = (SEQ / 32) * (DM / 16) * 32;    // 16384 per expert
    k_bfrag<<<dim3((thr2 + 255) / 256, NE), 256, 0, stream>>>(Ws, wsf, DM, SEQ);
    k_bfrag<<<dim3((thr2 + 255) / 256, NE), 256, 0, stream>>>(Wt, wtf, DM, SEQ);
  }

  k_gemm1<<<dim3(HID / 128, NROWS / 16), 256, 0, stream>>>(xs_h, gw1f, Hbuf);

  k_gate<<<(NROWS + 255) / 256, 256, 0, stream>>>(Hbuf, gw2, cnt, prow, pgat,
                                                  reidx, rgat);

  k_moe<<<dim3(DM / 128, NROWS / 16, NE), 256, 0, stream>>>(seas_h, tr_h, wsf,
                                                            wtf, cnt, prow,
                                                            pgat, y);

  k_loss<<<1, 256, 0, stream>>>(reidx, rgat, losscoef, y);
}
